// TransformerBlock_70231305224894
// MI455X (gfx1250) — matrix operand order repaired and checked by reading, not compiled, not run on hardware
//
#include <hip/hip_runtime.h>
#include <hip/hip_bf16.h>

typedef __bf16 v16bf __attribute__((ext_vector_type(16)));
typedef float  v8f   __attribute__((ext_vector_type(8)));

#define TOK    4096   // B*S
#define DMODEL 1024
#define DFFN   4096
#define SEQ    2048
#define HDIM   64

// ---------------------------------------------------------------------------
// CDNA5 async global->LDS copy (ASYNCcnt) helpers
// ---------------------------------------------------------------------------
__device__ __forceinline__ void async_load_b128(const void* lds_ptr, const void* gptr) {
    // low 32 bits of a flat LDS-aperture address are the LDS byte offset
    uint32_t lds_off = (uint32_t)(uintptr_t)lds_ptr;
    asm volatile("global_load_async_to_lds_b128 %0, %1, off"
                 :: "v"(lds_off), "v"((unsigned long long)(uintptr_t)gptr)
                 : "memory");
}
__device__ __forceinline__ void wait_async0() {
    asm volatile("s_wait_asynccnt 0" ::: "memory");
}

__device__ __forceinline__ float fast_gelu(float t) {
    // 0.5*t*(1+tanh(u)) == t*sigmoid(2u), u = 0.79788456*(t+0.044715 t^3)
    return t / (1.f + __expf(-1.5957691216f * (t + 0.044715f * t * t * t)));
}

// ---------------------------------------------------------------------------
// f32 -> bf16 conversion (weights)
// ---------------------------------------------------------------------------
__global__ __launch_bounds__(256) void cvt_bf16_kernel(const float* __restrict__ in,
                                                       __bf16* __restrict__ out, int n) {
    int i = blockIdx.x * 256 + threadIdx.x;
    if (i < n) out[i] = (__bf16)in[i];
}

// ---------------------------------------------------------------------------
// LayerNorm: one row (D=1024) per 256-thread block, bf16 output
// ---------------------------------------------------------------------------
__global__ __launch_bounds__(256) void ln_kernel(const float* __restrict__ x,
                                                 const float* __restrict__ gamma,
                                                 const float* __restrict__ beta,
                                                 __bf16* __restrict__ out) {
    __shared__ float r1[256], r2[256];
    int row = blockIdx.x, tid = threadIdx.x;
    const float* xr = x + (size_t)row * DMODEL;
    float lv[4], s = 0.f, ss = 0.f;
#pragma unroll
    for (int i = 0; i < 4; ++i) {
        int c = tid + 256 * i;
        lv[i] = xr[c];
        s += lv[i]; ss += lv[i] * lv[i];
    }
    r1[tid] = s; r2[tid] = ss;
    __syncthreads();
    for (int st = 128; st > 0; st >>= 1) {
        if (tid < st) { r1[tid] += r1[tid + st]; r2[tid] += r2[tid + st]; }
        __syncthreads();
    }
    float mean = r1[0] * (1.f / DMODEL);
    float var  = r2[0] * (1.f / DMODEL) - mean * mean;
    float inv  = rsqrtf(var + 1e-5f);
#pragma unroll
    for (int i = 0; i < 4; ++i) {
        int c = tid + 256 * i;
        out[(size_t)row * DMODEL + c] = (__bf16)((lv[i] - mean) * inv * gamma[c] + beta[c]);
    }
}

// ---------------------------------------------------------------------------
// WMMA bf16 GEMM: C = A[MxK] * B[KxN], block tile 128x128, BK=32,
// double-buffered LDS, one barrier per K-step, async prefetch overlapped
// with WMMA compute. 8 waves/block; each wave computes 32x64 (2x4 WMMA).
// MODE 1: scatter bf16 to [B,H,S,hd] head layout (QKV)
// MODE 2: outf = residual + C + bias (f32)
// MODE 3: outb = bf16(gelu(C + bias))
// ---------------------------------------------------------------------------
template<int MODE>
__global__ __launch_bounds__(256)
void gemm_bf16_kernel(const __bf16* __restrict__ A, const __bf16* __restrict__ B,
                      int N, int K,
                      const float* __restrict__ bias, const float* __restrict__ residual,
                      float* __restrict__ outf, __bf16* __restrict__ outb) {
    __shared__ __align__(16) __bf16 As[2][128][32];    // 16 KB, row-major
    __shared__ __align__(16) __bf16 Bst[2][128][32];   // 16 KB, transposed [col][k]

    int tid = threadIdx.x, lane = tid & 31, wave = tid >> 5;
    int wm = wave & 3, wn = wave >> 2;
    int half = (lane >> 4) & 1, l16 = lane & 15;
    int rowBase = blockIdx.y * 128, colBase = blockIdx.x * 128;

    auto load_tiles = [&](int k0, int buf) {
#pragma unroll
        for (int i = 0; i < 2; ++i) {               // A tile: async b128 -> LDS
            int idx = tid + 256 * i;
            int r = idx >> 2, cq = idx & 3;
            async_load_b128(&As[buf][r][cq * 8],
                            A + (size_t)(rowBase + r) * K + k0 + cq * 8);
        }
#pragma unroll
        for (int i = 0; i < 2; ++i) {               // B tile: load + transpose scatter
            int idx = tid + 256 * i;
            int r = idx >> 4, cg = idx & 15;
            uint4 w = *(const uint4*)(B + (size_t)(k0 + r) * N + colBase + cg * 8);
            __bf16 tmp[8];
            *(uint4*)tmp = w;
#pragma unroll
            for (int e = 0; e < 8; ++e)
                Bst[buf][cg * 8 + e][r] = tmp[e];
        }
    };

    v8f C[2][4] = {};
    const int kSteps = K >> 5;

    load_tiles(0, 0);
    for (int kt = 0; kt < kSteps; ++kt) {
        int cur = kt & 1;
        wait_async0();
        __syncthreads();                            // buffer `cur` ready
        if (kt + 1 < kSteps) load_tiles((kt + 1) << 5, cur ^ 1);

        v16bf a[2], bm[4];
#pragma unroll
        for (int i = 0; i < 2; ++i) {
            int r = 32 * wm + 16 * i + l16;
#pragma unroll
            for (int v = 0; v < 8; ++v) {
                int kk = half * 16 + 2 * v;
                a[i][2 * v]     = As[cur][r][kk];
                a[i][2 * v + 1] = As[cur][r][kk + 1];
            }
        }
#pragma unroll
        for (int j = 0; j < 4; ++j) {
            int c = 64 * wn + 16 * j + l16;
#pragma unroll
            for (int e = 0; e < 16; ++e)
                bm[j][e] = Bst[cur][c][half * 16 + e];   // contiguous b128 reads
        }
#pragma unroll
        for (int i = 0; i < 2; ++i)
#pragma unroll
            for (int j = 0; j < 4; ++j)
                C[i][j] = __builtin_amdgcn_wmma_f32_16x16x32_bf16(
                    false, a[i], false, bm[j], (short)0, C[i][j], false, false);
    }

    // ---- epilogue (compile-time mode) ----
    float bc[4];
    if constexpr (MODE == 2 || MODE == 3) {
#pragma unroll
        for (int j = 0; j < 4; ++j)
            bc[j] = bias[colBase + 64 * wn + 16 * j + l16];
    }
#pragma unroll
    for (int i = 0; i < 2; ++i) {
        int r0 = rowBase + 32 * wm + 16 * i + 8 * half;
#pragma unroll
        for (int j = 0; j < 4; ++j) {
            int c = colBase + 64 * wn + 16 * j + l16;
            if constexpr (MODE == 1) {
#pragma unroll
                for (int v = 0; v < 8; ++v) {
                    int r = r0 + v;
                    int bi = r >> 11, s = r & 2047, head = c >> 6, d = c & 63;
                    outb[(((size_t)bi * 16 + head) * SEQ + s) * HDIM + d] =
                        (__bf16)C[i][j][v];
                }
            } else if constexpr (MODE == 2) {
                float res[8];
#pragma unroll
                for (int v = 0; v < 8; ++v)
                    res[v] = residual[(size_t)(r0 + v) * N + c];
#pragma unroll
                for (int v = 0; v < 8; ++v)
                    outf[(size_t)(r0 + v) * N + c] = res[v] + C[i][j][v] + bc[j];
            } else {
#pragma unroll
                for (int v = 0; v < 8; ++v)
                    outb[(size_t)(r0 + v) * N + c] =
                        (__bf16)fast_gelu(C[i][j][v] + bc[j]);
            }
        }
    }
}

// ---------------------------------------------------------------------------
// Flash attention (causal): one (b*H+h, qblock) per 128-thread block.
// 4 waves; each wave owns 16 query rows. QK^T and PV via WMMA bf16.
// K/V tiles double-buffered: async prefetch of block jb+1 overlaps the
// softmax + WMMA work of block jb; one barrier per key block.
// ---------------------------------------------------------------------------
__global__ __launch_bounds__(128)
void attn_kernel(const __bf16* __restrict__ Qg, const __bf16* __restrict__ Kg,
                 const __bf16* __restrict__ Vg, __bf16* __restrict__ ctx) {
    __shared__ __align__(16) __bf16 Qs[64][64];
    __shared__ __align__(16) __bf16 Ks[2][64][64];
    __shared__ __align__(16) __bf16 Vst[2][64][64];   // transposed [hd][key]
    __shared__ __align__(16) __bf16 Ps[4][16][64];

    int bh = blockIdx.y, qb = blockIdx.x;
    int b = bh >> 4, head = bh & 15;
    const __bf16* Qh = Qg + (size_t)bh * SEQ * HDIM;
    const __bf16* Kh = Kg + (size_t)bh * SEQ * HDIM;
    const __bf16* Vh = Vg + (size_t)bh * SEQ * HDIM;

    int tid = threadIdx.x, lane = tid & 31, wave = tid >> 5;
    int half = (lane >> 4) & 1, l16 = lane & 15;

    auto load_kv = [&](int jb_, int buf) {
#pragma unroll
        for (int i = 0; i < 4; ++i) {               // K tile: async
            int idx = tid + 128 * i;
            int r = idx >> 3, cq = idx & 7;
            async_load_b128(&Ks[buf][r][cq * 8],
                            Kh + (size_t)(jb_ * 64 + r) * HDIM + cq * 8);
        }
#pragma unroll
        for (int i = 0; i < 4; ++i) {               // V tile: transpose scatter
            int idx = tid + 128 * i;
            int r = idx >> 3, cq = idx & 7;
            uint4 w = *(const uint4*)(Vh + (size_t)(jb_ * 64 + r) * HDIM + cq * 8);
            __bf16 tmp[8];
            *(uint4*)tmp = w;
#pragma unroll
            for (int e = 0; e < 8; ++e)
                Vst[buf][cq * 8 + e][r] = tmp[e];
        }
    };

#pragma unroll
    for (int i = 0; i < 4; ++i) {                   // Q tile 64x64, async
        int idx = tid + 128 * i;
        int r = idx >> 3, cq = idx & 7;
        async_load_b128(&Qs[r][cq * 8],
                        Qh + (size_t)(qb * 64 + r) * HDIM + cq * 8);
    }
    load_kv(0, 0);

    float m_i[8], l_i[8];
    v8f O[4] = {};
#pragma unroll
    for (int v = 0; v < 8; ++v) { m_i[v] = -1e30f; l_i[v] = 0.f; }

    for (int jb = 0; jb <= qb; ++jb) {
        int cur = jb & 1;
        wait_async0();
        __syncthreads();                            // buffer `cur` (and Qs) ready
        if (jb < qb) load_kv(jb + 1, cur ^ 1);

        // ---- S = Q K^T : wave's 16 rows x 64 keys ----
        v8f Sf[4] = {};
#pragma unroll
        for (int ks = 0; ks < 2; ++ks) {
            v16bf a;
            int qrow = wave * 16 + l16;
#pragma unroll
            for (int v = 0; v < 8; ++v) {
                int kk = ks * 32 + half * 16 + 2 * v;
                a[2 * v]     = Qs[qrow][kk];
                a[2 * v + 1] = Qs[qrow][kk + 1];
            }
#pragma unroll
            for (int t = 0; t < 4; ++t) {
                v16bf bm;
                int key = 16 * t + l16;
#pragma unroll
                for (int v = 0; v < 8; ++v) {
                    int kk = ks * 32 + half * 16 + 2 * v;
                    bm[2 * v]     = Ks[cur][key][kk];
                    bm[2 * v + 1] = Ks[cur][key][kk + 1];
                }
                Sf[t] = __builtin_amdgcn_wmma_f32_16x16x32_bf16(
                    false, a, false, bm, (short)0, Sf[t], false, false);
            }
        }

        // ---- online softmax ----
        float p[4][8];
#pragma unroll
        for (int t = 0; t < 4; ++t)
#pragma unroll
            for (int v = 0; v < 8; ++v) {
                float sv = Sf[t][v] * 0.125f;       // 1/sqrt(64)
                if (jb == qb) {
                    int key = jb * 64 + 16 * t + l16;
                    int qr  = qb * 64 + wave * 16 + v + 8 * half;
                    if (key > qr) sv = -1e30f;
                }
                p[t][v] = sv;
            }
        float mn[8], fac[8];
#pragma unroll
        for (int v = 0; v < 8; ++v) {
            float rm = fmaxf(fmaxf(p[0][v], p[1][v]), fmaxf(p[2][v], p[3][v]));
            rm = fmaxf(rm, __shfl_xor(rm, 1));
            rm = fmaxf(rm, __shfl_xor(rm, 2));
            rm = fmaxf(rm, __shfl_xor(rm, 4));
            rm = fmaxf(rm, __shfl_xor(rm, 8));
            mn[v]  = fmaxf(m_i[v], rm);
            fac[v] = __expf(m_i[v] - mn[v]);
            m_i[v] = mn[v];
        }
#pragma unroll
        for (int v = 0; v < 8; ++v) {
            float rs = 0.f;
#pragma unroll
            for (int t = 0; t < 4; ++t) {
                p[t][v] = __expf(p[t][v] - mn[v]);
                rs += p[t][v];
            }
            rs += __shfl_xor(rs, 1);
            rs += __shfl_xor(rs, 2);
            rs += __shfl_xor(rs, 4);
            rs += __shfl_xor(rs, 8);
            l_i[v] = l_i[v] * fac[v] + rs;
        }
#pragma unroll
        for (int t = 0; t < 4; ++t)
#pragma unroll
            for (int v = 0; v < 8; ++v) O[t][v] *= fac[v];

        // ---- transpose P through LDS (wave-private region) ----
#pragma unroll
        for (int t = 0; t < 4; ++t)
#pragma unroll
            for (int v = 0; v < 8; ++v)
                Ps[wave][v + 8 * half][16 * t + l16] = (__bf16)p[t][v];
        asm volatile("s_wait_dscnt 0" ::: "memory");

        // ---- O += P V ----
#pragma unroll
        for (int ks = 0; ks < 2; ++ks) {
            v16bf a;
#pragma unroll
            for (int v = 0; v < 8; ++v) {
                int kk = ks * 32 + half * 16 + 2 * v;
                a[2 * v]     = Ps[wave][l16][kk];
                a[2 * v + 1] = Ps[wave][l16][kk + 1];
            }
#pragma unroll
            for (int t = 0; t < 4; ++t) {
                v16bf bm;
#pragma unroll
                for (int e = 0; e < 16; ++e)
                    bm[e] = Vst[cur][16 * t + l16][ks * 32 + half * 16 + e];
                O[t] = __builtin_amdgcn_wmma_f32_16x16x32_bf16(
                    false, a, false, bm, (short)0, O[t], false, false);
            }
        }
    }

    // ---- normalize + write ctx in [token, emb] layout (bf16) ----
#pragma unroll
    for (int t = 0; t < 4; ++t)
#pragma unroll
        for (int v = 0; v < 8; ++v) {
            int srow = qb * 64 + wave * 16 + v + 8 * half;
            int tok  = b * SEQ + srow;
            int col  = head * HDIM + 16 * t + l16;
            ctx[(size_t)tok * DMODEL + col] = (__bf16)(O[t][v] / l_i[v]);
        }
}

// ---------------------------------------------------------------------------
// host launcher
// ---------------------------------------------------------------------------
extern "C" void kernel_launch(void* const* d_in, const int* in_sizes, int n_in,
                              void* d_out, int out_size, void* d_ws, size_t ws_size,
                              hipStream_t stream) {
    const float* x  = (const float*)d_in[0];
    const float* Wq = (const float*)d_in[1];
    const float* Wk = (const float*)d_in[2];
    const float* Wv = (const float*)d_in[3];
    const float* Wo = (const float*)d_in[4];
    const float* bo = (const float*)d_in[5];
    const float* W1 = (const float*)d_in[6];
    const float* b1 = (const float*)d_in[7];
    const float* W2 = (const float*)d_in[8];
    const float* b2 = (const float*)d_in[9];
    const float* g1 = (const float*)d_in[10];
    const float* s1 = (const float*)d_in[11];
    const float* g2 = (const float*)d_in[12];
    const float* s2 = (const float*)d_in[13];
    (void)in_sizes; (void)n_in; (void)out_size; (void)ws_size;

    char* ws = (char*)d_ws;
    size_t off = 0;
    auto take = [&](size_t bytes) -> char* {
        char* p = ws + off;
        off += (bytes + 255) & ~(size_t)255;
        return p;
    };
    __bf16* Wq_b  = (__bf16*)take((size_t)DMODEL * DMODEL * 2);
    __bf16* Wk_b  = (__bf16*)take((size_t)DMODEL * DMODEL * 2);
    __bf16* Wv_b  = (__bf16*)take((size_t)DMODEL * DMODEL * 2);
    __bf16* Wo_b  = (__bf16*)take((size_t)DMODEL * DMODEL * 2);
    __bf16* W1_b  = (__bf16*)take((size_t)DMODEL * DFFN * 2);
    __bf16* W2_b  = (__bf16*)take((size_t)DFFN * DMODEL * 2);
    __bf16* h_b   = (__bf16*)take((size_t)TOK * DMODEL * 2);
    __bf16* q_b   = (__bf16*)take((size_t)TOK * DMODEL * 2);
    __bf16* k_b   = (__bf16*)take((size_t)TOK * DMODEL * 2);
    __bf16* v_b   = (__bf16*)take((size_t)TOK * DMODEL * 2);
    __bf16* ctx_b = (__bf16*)take((size_t)TOK * DMODEL * 2);
    float*  x1    = (float*) take((size_t)TOK * DMODEL * 4);
    __bf16* h2_b  = (__bf16*)take((size_t)TOK * DMODEL * 2);
    __bf16* act_b = (__bf16*)take((size_t)TOK * DFFN * 2);

    const int nDD = DMODEL * DMODEL, nDF = DMODEL * DFFN;
    cvt_bf16_kernel<<<(nDD + 255) / 256, 256, 0, stream>>>(Wq, Wq_b, nDD);
    cvt_bf16_kernel<<<(nDD + 255) / 256, 256, 0, stream>>>(Wk, Wk_b, nDD);
    cvt_bf16_kernel<<<(nDD + 255) / 256, 256, 0, stream>>>(Wv, Wv_b, nDD);
    cvt_bf16_kernel<<<(nDD + 255) / 256, 256, 0, stream>>>(Wo, Wo_b, nDD);
    cvt_bf16_kernel<<<(nDF + 255) / 256, 256, 0, stream>>>(W1, W1_b, nDF);
    cvt_bf16_kernel<<<(nDF + 255) / 256, 256, 0, stream>>>(W2, W2_b, nDF);

    // LN1
    ln_kernel<<<TOK, 256, 0, stream>>>(x, g1, s1, h_b);

    // QKV projections (scatter into [B,H,S,hd])
    dim3 gProj(DMODEL / 128, TOK / 128);
    gemm_bf16_kernel<1><<<gProj, 256, 0, stream>>>(h_b, Wq_b, DMODEL, DMODEL,
                                                   nullptr, nullptr, nullptr, q_b);
    gemm_bf16_kernel<1><<<gProj, 256, 0, stream>>>(h_b, Wk_b, DMODEL, DMODEL,
                                                   nullptr, nullptr, nullptr, k_b);
    gemm_bf16_kernel<1><<<gProj, 256, 0, stream>>>(h_b, Wv_b, DMODEL, DMODEL,
                                                   nullptr, nullptr, nullptr, v_b);

    // causal flash attention
    attn_kernel<<<dim3(SEQ / 64, 2 * 16), 128, 0, stream>>>(q_b, k_b, v_b, ctx_b);

    // output projection + bias + residual -> x1 (f32)
    gemm_bf16_kernel<2><<<gProj, 256, 0, stream>>>(ctx_b, Wo_b, DMODEL, DMODEL,
                                                   bo, x, x1, nullptr);

    // LN2
    ln_kernel<<<TOK, 256, 0, stream>>>(x1, g2, s2, h2_b);

    // FFN1: gelu(h2 @ W1 + b1) -> bf16
    gemm_bf16_kernel<3><<<dim3(DFFN / 128, TOK / 128), 256, 0, stream>>>(
        h2_b, W1_b, DFFN, DMODEL, b1, nullptr, nullptr, act_b);

    // FFN2: x1 + act @ W2 + b2 -> d_out (f32)
    gemm_bf16_kernel<2><<<gProj, 256, 0, stream>>>(act_b, W2_b, DMODEL, DFFN,
                                                   b2, x1, (float*)d_out, nullptr);
}
